// GRADE_65154653880994
// MI455X (gfx1250) — compile-verified, matrix operand order checked by reading, not run on hardware
//
#include <hip/hip_runtime.h>
#include <hip/hip_bf16.h>

// ---------- types ----------
typedef __attribute__((ext_vector_type(16))) __bf16 bf16x16;
typedef __attribute__((ext_vector_type(8)))  float  f32x8;

#define B_  2048
#define N_  10
#define D_  300
#define FH_ 75
#define H_  4
#define L_  3

__device__ __forceinline__ void pack8(bf16x16& dst, int off, float4 v0, float4 v1)
{
    dst[off + 0] = (__bf16)v0.x; dst[off + 1] = (__bf16)v0.y;
    dst[off + 2] = (__bf16)v0.z; dst[off + 3] = (__bf16)v0.w;
    dst[off + 4] = (__bf16)v1.x; dst[off + 5] = (__bf16)v1.y;
    dst[off + 6] = (__bf16)v1.z; dst[off + 7] = (__bf16)v1.w;
}

// ---------------------------------------------------------------------------
// WMMA GEMM:  C[M,N] = act( A[M,K] * W[N,K]^T + bias )
// A row-major lda, W row-major ldb (both contiguous along K), C row-major ldc.
// REQUIRES lda % 4 == 0 and ldb % 4 == 0 (16B-aligned fragment runs).
// One wave computes a 32x16 output tile (2 A-frags share 1 B-frag);
// 8 waves / block. act: 0 = none, 1 = ELU.
// Out-of-range rows/cols are handled by pointer clamping (their outputs are
// simply never stored); only the K-tail is masked (single iteration,
// clamped unconditional loads + select-to-zero -> no exec divergence).
// ---------------------------------------------------------------------------
__global__ __launch_bounds__(256) void wmma_gemm_bf16(
    const float* __restrict__ A, int lda,
    const float* __restrict__ W, int ldb,
    const float* __restrict__ bias,
    float* __restrict__ C, int ldc,
    int M, int N, int K, int act)
{
    const int wave = threadIdx.x >> 5;
    const int lane = threadIdx.x & 31;
    const int half = lane >> 4;       // which K-half this lane holds
    const int r    = lane & 15;       // row within fragment

    const int tilesN = (N + 15) >> 4;
    const int tilesM = (M + 31) >> 5;                // 32 rows per wave
    const int tile   = blockIdx.x * 8 + wave;        // wave-uniform
    if (tile >= tilesM * tilesN) return;             // wave-uniform exit
    const int tm = tile / tilesN;
    const int tn = tile % tilesN;

    const int am0 = tm * 32 + r;
    const int am1 = am0 + 16;
    const int bn  = tn * 16 + r;
    const float* __restrict__ arow0 = A + (size_t)(am0 < M ? am0 : M - 1) * lda;
    const float* __restrict__ arow1 = A + (size_t)(am1 < M ? am1 : M - 1) * lda;
    const float* __restrict__ brow  = W + (size_t)(bn  < N ? bn  : N - 1) * ldb;

    f32x8 acc0 = {}, acc1 = {};
    const int Kfull = K & ~31;
    int k0 = 0;
    // main loop: unconditional 16B vector loads
    for (; k0 < Kfull; k0 += 32) {
        const int b0 = k0 + half * 8;        // frag elements j = 0..7  (K b0..b0+7)
        const int b1 = b0 + 16;              // frag elements j = 8..15 (K b1..b1+7)
        const float4 a00 = *(const float4*)(arow0 + b0);
        const float4 a01 = *(const float4*)(arow0 + b0 + 4);
        const float4 a02 = *(const float4*)(arow0 + b1);
        const float4 a03 = *(const float4*)(arow0 + b1 + 4);
        const float4 a10 = *(const float4*)(arow1 + b0);
        const float4 a11 = *(const float4*)(arow1 + b0 + 4);
        const float4 a12 = *(const float4*)(arow1 + b1);
        const float4 a13 = *(const float4*)(arow1 + b1 + 4);
        const float4 w0  = *(const float4*)(brow  + b0);
        const float4 w1  = *(const float4*)(brow  + b0 + 4);
        const float4 w2  = *(const float4*)(brow  + b1);
        const float4 w3  = *(const float4*)(brow  + b1 + 4);
        bf16x16 af0, af1, bf;
        pack8(af0, 0, a00, a01); pack8(af0, 8, a02, a03);
        pack8(af1, 0, a10, a11); pack8(af1, 8, a12, a13);
        pack8(bf,  0, w0,  w1);  pack8(bf,  8, w2,  w3);
        acc0 = __builtin_amdgcn_wmma_f32_16x16x32_bf16(
                   false, af0, false, bf, (short)0, acc0, false, false);
        acc1 = __builtin_amdgcn_wmma_f32_16x16x32_bf16(
                   false, af1, false, bf, (short)0, acc1, false, false);
    }
    // K tail: one masked chunk, clamped unconditional loads + select
    if (k0 < K) {
        bf16x16 af0, af1, bf;
        #pragma unroll
        for (int j = 0; j < 16; ++j) {
            const int k  = k0 + half * 8 + j + ((j >= 8) ? 8 : 0);
            const int kc = (k < K) ? k : (K - 1);
            float a0 = arow0[kc];
            float a1 = arow1[kc];
            float wv = brow[kc];
            a0 = (k < K) ? a0 : 0.0f;
            a1 = (k < K) ? a1 : 0.0f;
            wv = (k < K) ? wv : 0.0f;
            af0[j] = (__bf16)a0;
            af1[j] = (__bf16)a1;
            bf[j]  = (__bf16)wv;
        }
        acc0 = __builtin_amdgcn_wmma_f32_16x16x32_bf16(
                   false, af0, false, bf, (short)0, acc0, false, false);
        acc1 = __builtin_amdgcn_wmma_f32_16x16x32_bf16(
                   false, af1, false, bf, (short)0, acc1, false, false);
    }

    // C/D layout: VGPR v, lanes 0-15 -> row v, lanes 16-31 -> row v+8; col = r
    const int col = tn * 16 + r;
    if (col < N) {
        const float bv = bias ? bias[col] : 0.0f;
        #pragma unroll
        for (int v = 0; v < 8; ++v) {
            const int row0 = tm * 32 + v + 8 * half;
            const int row1 = row0 + 16;
            if (row0 < M) {
                float x = acc0[v] + bv;
                if (act == 1) x = (x > 0.0f) ? x : (__expf(x) - 1.0f);
                C[(size_t)row0 * ldc + col] = x;
            }
            if (row1 < M) {
                float x = acc1[v] + bv;
                if (act == 1) x = (x > 0.0f) ? x : (__expf(x) - 1.0f);
                C[(size_t)row1 * ldc + col] = x;
            }
        }
    }
}

// ---------------------------------------------------------------------------
// Embedding gather: kw[b,n,:] = emb[ids[b,n],:]
// ---------------------------------------------------------------------------
__global__ void gather_kw(const float* __restrict__ emb,
                          const int* __restrict__ ids,
                          float* __restrict__ kw, int total)
{
    int i = blockIdx.x * blockDim.x + threadIdx.x;
    if (i >= total) return;
    int bn = i / D_, f = i - bn * D_;
    kw[i] = emb[(size_t)ids[bn] * D_ + f];
}

__global__ void count_nz(const int* __restrict__ ids, float* __restrict__ nz, int B)
{
    int b = blockIdx.x * blockDim.x + threadIdx.x;
    if (b >= B) return;
    int c = 0;
    #pragma unroll
    for (int n = 0; n < N_; ++n) c += (ids[b * N_ + n] != 0) ? 1 : 0;
    nz[b] = (float)c;
}

// ---------------------------------------------------------------------------
// GAT attention for one layer, one block per batch element, all 4 heads.
// Closed form of the repeat/cat/view construction:
//   p<5 : e = d1[2p+(q>=5)] + d2[2p+(q>=5)]
//   p>=5: e = d1[2(q%5)]    + d2[2(q%5)+1]
// raw = adj ? leaky_relu(e) : 0 ; softmax over p ; z = att-weighted h.
// ---------------------------------------------------------------------------
__global__ __launch_bounds__(256) void gat_attention(
    const float* __restrict__ hproj,   // [B*N, 300]
    const float* __restrict__ a_l,     // [H, 150]
    const int*   __restrict__ adjs,    // [B, N, N]
    float* __restrict__ z)             // [B*N, 300]
{
    __shared__ float sh_h[N_ * D_];          // 3000
    __shared__ float sh_d1[H_ * N_];
    __shared__ float sh_d2[H_ * N_];
    __shared__ float sh_att[H_ * N_ * N_];   // 400

    const int b   = blockIdx.x;
    const int tid = threadIdx.x;

    for (int i = tid; i < N_ * D_; i += 256)
        sh_h[i] = hproj[(size_t)b * N_ * D_ + i];
    __syncthreads();

    if (tid < 2 * H_ * N_) {                 // 80 dot products of length 75
        const int h = tid / (2 * N_);
        const int rem = tid % (2 * N_);
        const int n = rem >> 1, w = rem & 1;
        const float* av = a_l + h * 2 * FH_ + w * FH_;
        const float* hv = sh_h + n * D_ + h * FH_;
        float s = 0.f;
        #pragma unroll 5
        for (int f = 0; f < FH_; ++f) s += hv[f] * av[f];
        if (w == 0) sh_d1[h * N_ + n] = s; else sh_d2[h * N_ + n] = s;
    }
    __syncthreads();

    if (tid < H_ * N_) {                     // one (head, q) column per thread
        const int h = tid / N_, q = tid % N_;
        float raw[N_];
        #pragma unroll
        for (int p = 0; p < N_; ++p) {
            float e;
            if (p < 5) {
                const int rr = 2 * p + ((q >= 5) ? 1 : 0);
                e = sh_d1[h * N_ + rr] + sh_d2[h * N_ + rr];
            } else {
                const int rr = 2 * (q % 5);
                e = sh_d1[h * N_ + rr] + sh_d2[h * N_ + rr + 1];
            }
            e = (e > 0.f) ? e : 0.2f * e;                         // leaky relu
            const int adj = adjs[b * N_ * N_ + p * N_ + q];
            raw[p] = (adj > 0) ? e : 0.0f;                        // (-9e15)*0 == -0
        }
        float mx = raw[0];
        #pragma unroll
        for (int p = 1; p < N_; ++p) mx = fmaxf(mx, raw[p]);
        float ex[N_], den = 0.f;
        #pragma unroll
        for (int p = 0; p < N_; ++p) { ex[p] = __expf(raw[p] - mx); den += ex[p]; }
        const float inv = 1.0f / den;
        #pragma unroll
        for (int p = 0; p < N_; ++p) sh_att[h * 100 + p * N_ + q] = ex[p] * inv;
    }
    __syncthreads();

    for (int t = tid; t < N_ * D_; t += 256) {   // z[b,p, h*75+f]
        const int h = t / (N_ * FH_);
        const int rem = t % (N_ * FH_);
        const int p = rem / FH_, f = rem % FH_;
        float acc = 0.f;
        #pragma unroll
        for (int q = 0; q < N_; ++q)
            acc += sh_att[h * 100 + p * N_ + q] * sh_h[q * D_ + h * FH_ + f];
        z[(size_t)b * N_ * D_ + p * D_ + h * FH_ + f] = acc;
    }
}

// h_next = elu(lin[b,:] (broadcast over nodes) + z)
__global__ void fuse_elu(const float* __restrict__ lin,
                         const float* __restrict__ z,
                         float* __restrict__ hout, int total)
{
    int i = blockIdx.x * blockDim.x + threadIdx.x;
    if (i >= total) return;
    const int b = i / (N_ * D_);
    const int c = i % D_;
    const float x = lin[b * D_ + c] + z[i];
    hout[i] = (x > 0.f) ? x : (__expf(x) - 1.f);
}

// pooled_in[b,f] = sum_n h[b,n,f] / nz[b]
__global__ void pool_mean(const float* __restrict__ h,
                          const float* __restrict__ nz,
                          float* __restrict__ out, int total)
{
    int i = blockIdx.x * blockDim.x + threadIdx.x;
    if (i >= total) return;
    const int b = i / D_, f = i % D_;
    float s = 0.f;
    #pragma unroll
    for (int n = 0; n < N_; ++n) s += h[(size_t)b * N_ * D_ + n * D_ + f];
    out[i] = s / nz[b];
}

// score[b] = sigmoid(dot(lin4[b], W5) + b5)
__global__ void final_score(const float* __restrict__ lin4,
                            const float* __restrict__ W5,
                            const float* __restrict__ b5,
                            float* __restrict__ out, int B)
{
    int b = blockIdx.x * blockDim.x + threadIdx.x;
    if (b >= B) return;
    float s = b5[0];
    #pragma unroll 8
    for (int k = 0; k < 128; ++k) s += lin4[b * 128 + k] * W5[k];
    out[b] = 1.0f / (1.0f + __expf(-s));
}

// ---------------------------------------------------------------------------
extern "C" void kernel_launch(void* const* d_in, const int* in_sizes, int n_in,
                              void* d_out, int out_size, void* d_ws, size_t ws_size,
                              hipStream_t stream)
{
    const float* emb      = (const float*)d_in[0];
    const float* gat_W    = (const float*)d_in[1];   // [L,H,FH,D]
    const float* gat_a    = (const float*)d_in[2];   // [L,H,2FH]
    const float* W0       = (const float*)d_in[3];   // [L,300,3000]
    const float* b0       = (const float*)d_in[4];
    const float* W1       = (const float*)d_in[5];   // [512,300]
    const float* b1       = (const float*)d_in[6];
    const float* W2       = (const float*)d_in[7];   // [512,768]
    const float* b2       = (const float*)d_in[8];
    const float* W3       = (const float*)d_in[9];   // [512,1024]
    const float* b3       = (const float*)d_in[10];
    const float* W4       = (const float*)d_in[11];  // [128,512]
    const float* b4       = (const float*)d_in[12];
    const float* W5       = (const float*)d_in[13];  // [1,128]
    const float* b5       = (const float*)d_in[14];
    const float* bert_cls = (const float*)d_in[15];  // [B,768]
    const int*   kw_ids   = (const int*)d_in[16];    // [B,N]
    const int*   adjs     = (const int*)d_in[17];    // [B,N,N]
    float* score = (float*)d_out;

    // ---- workspace layout (floats) ----
    float* ws = (float*)d_ws;
    const size_t SZ_H = (size_t)B_ * N_ * D_;        // 6,144,000
    float* hA    = ws;
    float* hB    = hA + SZ_H;
    float* hproj = hB + SZ_H;
    float* zbuf  = hproj + SZ_H;
    float* lin   = zbuf + SZ_H;                      // [B,300]
    float* nz    = lin + (size_t)B_ * D_;            // [B]
    // tail reuses freed layer buffers
    float* pooled_in = lin;                          // [B,300]
    float* fusin     = hproj;                        // [B,1024] (bert|pooled)
    float* fus       = zbuf;                         // [B,512]
    float* lin4      = zbuf + (size_t)B_ * 512;      // [B,128]

    const int TPB = 256;
    const int elemH = (int)SZ_H;                     // 6,144,000

    // 1) gather + nz
    gather_kw<<<(elemH + TPB - 1) / TPB, TPB, 0, stream>>>(emb, kw_ids, hA, elemH);
    count_nz<<<(B_ + TPB - 1) / TPB, TPB, 0, stream>>>(kw_ids, nz, B_);

    auto gemm = [&](const float* Ap, int lda, const float* Wt, int ldb,
                    const float* bias, float* Cp, int ldc,
                    int M, int N, int K, int act) {
        const int tiles = ((M + 31) / 32) * ((N + 15) / 16);
        wmma_gemm_bf16<<<(tiles + 7) / 8, 256, 0, stream>>>(
            Ap, lda, Wt, ldb, bias, Cp, ldc, M, N, K, act);
    };

    // 2) three GAT layers
    float* hcur = hA;
    float* hnext = hB;
    for (int l = 0; l < L_; ++l) {
        // projection: [B*N,300] x [300,300]^T -> hproj (cols = h*75+f)
        gemm(hcur, D_, gat_W + (size_t)l * H_ * FH_ * D_, D_,
             nullptr, hproj, D_, B_ * N_, H_ * FH_, D_, 0);
        // attention -> zbuf
        gat_attention<<<B_, 256, 0, stream>>>(hproj, gat_a + (size_t)l * H_ * 2 * FH_,
                                              adjs, zbuf);
        // W0 linear: [B,3000] x [300,3000]^T -> lin [B,300]
        gemm(hcur, N_ * D_, W0 + (size_t)l * D_ * N_ * D_, N_ * D_,
             b0 + (size_t)l * D_, lin, D_, B_, D_, N_ * D_, 0);
        // h_next = elu(lin + z)
        fuse_elu<<<(elemH + TPB - 1) / TPB, TPB, 0, stream>>>(lin, zbuf, hnext, elemH);
        float* t = hcur; hcur = hnext; hnext = t;
    }

    // 3) mean-pool over nodes / nz
    pool_mean<<<((B_ * D_) + TPB - 1) / TPB, TPB, 0, stream>>>(hcur, nz, pooled_in, B_ * D_);

    // 4) bert branch -> fusin[:, 0:512]  (no activation)
    gemm(bert_cls, 768, W2, 768, b2, fusin + 0, 1024, B_, 512, 768, 0);
    // 5) pooled branch -> fusin[:, 512:1024]  (elu)
    gemm(pooled_in, D_, W1, D_, b1, fusin + 512, 1024, B_, 512, D_, 1);
    // 6) fusion: [B,1024] x [512,1024]^T -> fus (elu)
    gemm(fusin, 1024, W3, 1024, b3, fus, 512, B_, 512, 1024, 1);
    // 7) lin4: [B,512] x [128,512]^T -> lin4 (elu)
    gemm(fus, 512, W4, 512, b4, lin4, 128, B_, 128, 512, 1);
    // 8) sigmoid head
    final_score<<<(B_ + TPB - 1) / TPB, TPB, 0, stream>>>(lin4, W5, b5, score, B_);
}
